// VAE_31439160607270
// MI455X (gfx1250) — matrix operand order repaired and checked by reading, not compiled, not run on hardware
//
#include <hip/hip_runtime.h>
#include <hip/hip_bf16.h>
#include <math.h>
#include <stdint.h>

typedef __attribute__((ext_vector_type(16))) _Float16 v16h;
typedef __attribute__((ext_vector_type(8)))  _Float16 v8h;
typedef __attribute__((ext_vector_type(4)))  _Float16 v4h;
typedef __attribute__((ext_vector_type(8)))  float    v8f;
typedef __attribute__((ext_vector_type(4)))  float    v4f;

#define BATCH 65536
#define DATA  784
#define HID   500
#define HIDP  512
#define LAT   32
#define KP1   800   /* DATA padded to multiple of 32 */

/* ---- workspace layout (bytes) ---- */
static const size_t OFF_H    = 0;                        /* h / hd  f16 [B,512]  67108864 */
static const size_t OFF_Z    = 67108864;                 /* z  f16 [B,32]         4194304 */
static const size_t OFF_R1   = 71303168;                 /* r1 f32 [B]             262144 */
static const size_t OFF_W1P  = 71565312;                 /* 800x512 f16            819200 */
static const size_t OFF_WMUP = 72384512;                 /* 512x32 f16              32768 */
static const size_t OFF_WSGP = 72417280;                 /* 512x32 f16              32768 */
static const size_t OFF_WD1P = 72450048;                 /* 32x512 f16              32768 */
static const size_t OFF_WD2P = 72482816;                 /* 512x784 f16            802816 */

static __device__ __forceinline__ v8f wmma16(v16h a, v16h b, v8f c) {
  return __builtin_amdgcn_wmma_f32_16x16x32_f16(false, a, false, b, (short)0, c, false, false);
}

/* A-fragment (16x32 f16) from row-major f16 (global or LDS), p = row_base + k0 + kb */
static __device__ __forceinline__ v16h load_a16(const _Float16* p) {
  v8h lo = *(const v8h*)p;
  v8h hi = *(const v8h*)(p + 16);
  v16h a;
#pragma unroll
  for (int i = 0; i < 8; ++i) { a[i] = lo[i]; a[i + 8] = hi[i]; }
  return a;
}

static __device__ __forceinline__ void wait_asynccnt0() {
#if __has_builtin(__builtin_amdgcn_s_wait_asynccnt)
  __builtin_amdgcn_s_wait_asynccnt(0);
#else
  asm volatile("s_wait_asynccnt 0" ::: "memory");
#endif
}

/* ---------- pack f32 weight [K,N] row-major -> f16 B-fragments [nt][kt][lane][16] ---------- */
__global__ __launch_bounds__(256) void k_pack(const float* __restrict__ W, _Float16* __restrict__ dst,
                                              int K, int N, int Kp, int Np) {
  int t = blockIdx.x * 256 + threadIdx.x;
  int total = Kp * Np;
  if (t >= total) return;
  int fragIdx = t >> 9;
  int within  = t & 511;
  int lane = within >> 4;
  int j    = within & 15;
  int KT = Kp >> 5;
  int nt = fragIdx / KT;
  int kt = fragIdx - nt * KT;
  int n = nt * 16 + (lane & 15);
  int k = kt * 32 + ((lane >> 4) << 3) + (j & 7) + ((j >> 3) << 4);
  float v = (k < K && n < N) ? W[(size_t)k * N + n] : 0.0f;
  dst[t] = (_Float16)v;
}

__global__ void k_zero_out(float* out) { out[0] = 0.0f; }

/* ---------- GEMM1: h = tanh(x @ W1 + b1), x f32 [B,784] -> h f16 [B,512]
   Block: 32 M rows x full N=512. x tile converted once into LDS; each wave
   covers 64 N cols x 2 M-subtiles so every B fragment feeds 2 WMMAs. ---------- */
__global__ __launch_bounds__(256) void k_gemm1(const float* __restrict__ x, const _Float16* __restrict__ W1p,
                                               const float* __restrict__ b1, _Float16* __restrict__ h) {
  __shared__ _Float16 Atile[32 * KP1];    /* 50 KB */
  const int tid  = threadIdx.x;
  const int lane = tid & 31;
  const int w    = tid >> 5;
  const int m0   = blockIdx.x * 32;
  const int n0   = w * 64;
  const int kb   = (lane >> 4) * 8;
  const int KT   = KP1 / 32; /* 25 */

  /* ---- stage & convert x tile: 8 threads per row, 100 cols each ---- */
  {
    const int r  = tid >> 3;               /* 0..31 */
    const int c0 = (tid & 7) * 100;        /* 0..700 */
    const float* xr = x + (size_t)(m0 + r) * DATA;
    _Float16* dr = &Atile[r * KP1 + c0];
#pragma unroll
    for (int i = 0; i < 25; ++i) {
      int c = c0 + i * 4;
      v4h o;
      if (c + 4 <= DATA) {
        v4f f = *(const v4f*)(xr + c);
#pragma unroll
        for (int q = 0; q < 4; ++q) o[q] = (_Float16)f[q];
      } else {
#pragma unroll
        for (int q = 0; q < 4; ++q) o[q] = (_Float16)0.0f;
      }
      *(v4h*)(dr + i * 4) = o;
    }
  }
  __syncthreads();

  v8f acc[2][4];
#pragma unroll
  for (int s = 0; s < 2; ++s)
#pragma unroll
    for (int f = 0; f < 4; ++f)
#pragma unroll
      for (int i = 0; i < 8; ++i) acc[s][f][i] = 0.0f;

  for (int kt = 0; kt < KT; ++kt) {
    const int k0 = kt * 32;
    v16h a0 = load_a16(&Atile[((lane & 15)) * KP1 + k0 + kb]);
    v16h a1 = load_a16(&Atile[(16 + (lane & 15)) * KP1 + k0 + kb]);
#pragma unroll
    for (int f = 0; f < 4; ++f) {
      const v16h b = *(const v16h*)(W1p + (size_t)(((n0 >> 4) + f) * KT + kt) * 512 + lane * 16);
      acc[0][f] = wmma16(a0, b, acc[0][f]);
      acc[1][f] = wmma16(a1, b, acc[1][f]);
    }
  }
  const int rb = (lane >> 4) * 8;
  const int cb = lane & 15;
#pragma unroll
  for (int s = 0; s < 2; ++s)
#pragma unroll
    for (int f = 0; f < 4; ++f) {
      int n = n0 + f * 16 + cb;
      float bias = (n < HID) ? b1[n] : 0.0f;
#pragma unroll
      for (int j = 0; j < 8; ++j) {
        float v = tanhf(acc[s][f][j] + bias);
        h[(size_t)(m0 + s * 16 + rb + j) * HIDP + n] = (_Float16)v;
      }
    }
}

/* ---------- GEMM2/3 fused: mu, logsig, z, r1 = 0.5*sum(ls + eps^2 - z^2) ---------- */
__global__ __launch_bounds__(256) void k_gemm23(const _Float16* __restrict__ h,
                                                const _Float16* __restrict__ Wmup,
                                                const _Float16* __restrict__ Wsgp,
                                                const float* __restrict__ bmu, const float* __restrict__ bsg,
                                                const float* __restrict__ eps,
                                                _Float16* __restrict__ z, float* __restrict__ r1) {
  const int lane = threadIdx.x & 31;
  const int w    = threadIdx.x >> 5;
  const int m0   = blockIdx.x * 128 + w * 16;
  const int row  = m0 + (lane & 15);
  const int kb   = (lane >> 4) * 8;
  const int KT   = HIDP / 32; /* 16 */

  v8f amu[2], asg[2];
#pragma unroll
  for (int f = 0; f < 2; ++f)
#pragma unroll
    for (int i = 0; i < 8; ++i) { amu[f][i] = 0.0f; asg[f][i] = 0.0f; }

  const _Float16* hrow = h + (size_t)row * HIDP;
  for (int kt = 0; kt < KT; ++kt) {
    v16h a = load_a16(hrow + kt * 32 + kb);
#pragma unroll
    for (int f = 0; f < 2; ++f) {
      const v16h bm = *(const v16h*)(Wmup + (size_t)(f * KT + kt) * 512 + lane * 16);
      const v16h bs = *(const v16h*)(Wsgp + (size_t)(f * KT + kt) * 512 + lane * 16);
      amu[f] = wmma16(a, bm, amu[f]);
      asg[f] = wmma16(a, bs, asg[f]);
    }
  }
  const int rb = (lane >> 4) * 8;
  const int cb = lane & 15;
  float tsum[8];
#pragma unroll
  for (int j = 0; j < 8; ++j) tsum[j] = 0.0f;
#pragma unroll
  for (int f = 0; f < 2; ++f) {
    int col = cb + f * 16;
    float bm = bmu[col], bs = bsg[col];
#pragma unroll
    for (int j = 0; j < 8; ++j) {
      int r = m0 + rb + j;
      float mu = amu[f][j] + bm;
      float ls = asg[f][j] + bs;
      float e  = eps[(size_t)r * LAT + col];
      float zz = mu + e * __expf(0.5f * ls); /* mu + eps*sqrt(exp(ls)) */
      z[(size_t)r * LAT + col] = (_Float16)zz;
      tsum[j] += ls + e * e - zz * zz;      /* 2*(log p_z - log q) per element */
    }
  }
#pragma unroll
  for (int j = 0; j < 8; ++j) {
    float t = tsum[j];
#pragma unroll
    for (int m = 1; m < 16; m <<= 1) t += __shfl_xor(t, m, 32);
    if ((lane & 15) == 0) r1[m0 + rb + j] = 0.5f * t;
  }
}

/* ---------- GEMM4: hd = tanh(z @ Wd1 + bd1), single K=32 chunk ---------- */
__global__ __launch_bounds__(256) void k_gemm4(const _Float16* __restrict__ z, const _Float16* __restrict__ Wd1p,
                                               const float* __restrict__ bd1, _Float16* __restrict__ hd) {
  const int lane = threadIdx.x & 31;
  const int w    = threadIdx.x >> 5;
  const int m0   = blockIdx.x * 16;
  const int n0   = w * 64;
  const int row  = m0 + (lane & 15);
  const int kb   = (lane >> 4) * 8;

  v16h a = load_a16(z + (size_t)row * LAT + kb);
  const int rb = m0 + (lane >> 4) * 8;
#pragma unroll
  for (int f = 0; f < 4; ++f) {
    v8f acc;
#pragma unroll
    for (int i = 0; i < 8; ++i) acc[i] = 0.0f;
    const v16h b = *(const v16h*)(Wd1p + (size_t)((n0 >> 4) + f) * 512 + lane * 16);
    acc = wmma16(a, b, acc);
    int n = n0 + f * 16 + (lane & 15);
    float bias = (n < HID) ? bd1[n] : 0.0f;
#pragma unroll
    for (int j = 0; j < 8; ++j) {
      float v = tanhf(acc[j] + bias);
      hd[(size_t)(rb + j) * HIDP + n] = (_Float16)v;
    }
  }
}

/* ---------- GEMM5 + Bernoulli log-lik + final ELBO reduce.
   Block: 32 M rows x all 49 N tiles. hd tile staged to LDS with CDNA5
   async global->LDS copies; waves read A fragments from LDS. ---------- */
__global__ __launch_bounds__(256) void k_gemm5(const _Float16* __restrict__ hd, const _Float16* __restrict__ Wd2p,
                                               const float* __restrict__ bd2, const float* __restrict__ x,
                                               const float* __restrict__ r1, float* __restrict__ out) {
  __shared__ _Float16 Ahd[32 * HIDP];     /* 32 KB */
  __shared__ float rowsum[32];
  const int tid  = threadIdx.x;
  const int lane = tid & 31;
  const int w    = tid >> 5;
  const int m0   = blockIdx.x * 32;
  const int kb   = (lane >> 4) * 8;
  const int KT   = HIDP / 32; /* 16 */
  const int NT   = DATA / 16; /* 49 */

  /* ---- async-stage hd tile (32x512 f16 = 32 KB): 8 x b128 per thread ---- */
  {
    uint32_t lds0 = (uint32_t)(uintptr_t)(&Ahd[0]);
    uint64_t ga0  = (uint64_t)(uintptr_t)(hd + (size_t)m0 * HIDP);
#pragma unroll
    for (int i = 0; i < 8; ++i) {
      uint32_t loff = lds0 + (uint32_t)(tid * 16 + i * 4096);
      uint64_t ga   = ga0 + (uint64_t)(tid * 16 + i * 4096);
      asm volatile("global_load_async_to_lds_b128 %0, %1, off"
                   :: "v"(loff), "v"(ga) : "memory");
    }
  }
  if (tid < 32) rowsum[tid] = 0.0f;
  wait_asynccnt0();
  __syncthreads();

  const int rb = (lane >> 4) * 8;
  const int cb = lane & 15;
  float psum[2][8];
#pragma unroll
  for (int s = 0; s < 2; ++s)
#pragma unroll
    for (int j = 0; j < 8; ++j) psum[s][j] = 0.0f;

  for (int nt = w; nt < NT; nt += 8) {   /* wave-uniform loop: EXEC stays full */
    v8f acc[2];
#pragma unroll
    for (int s = 0; s < 2; ++s)
#pragma unroll
      for (int i = 0; i < 8; ++i) acc[s][i] = 0.0f;
    for (int kt = 0; kt < KT; ++kt) {
      const v16h b = *(const v16h*)(Wd2p + (size_t)(nt * KT + kt) * 512 + lane * 16);
      v16h a0 = load_a16(&Ahd[((lane & 15)) * HIDP + kt * 32 + kb]);
      v16h a1 = load_a16(&Ahd[(16 + (lane & 15)) * HIDP + kt * 32 + kb]);
      acc[0] = wmma16(a0, b, acc[0]);
      acc[1] = wmma16(a1, b, acc[1]);
    }
    int n = nt * 16 + cb;
    float bias = bd2[n];
#pragma unroll
    for (int s = 0; s < 2; ++s)
#pragma unroll
      for (int j = 0; j < 8; ++j) {
        int r = m0 + s * 16 + rb + j;
        float av = acc[s][j] + bias;
        float xv = x[(size_t)r * DATA + n];
        /* x*log(p)+(1-x)*log(1-p) = x*a - softplus(a), stable softplus */
        float sp = (av > 0.0f ? av : 0.0f) + log1pf(__expf(-fabsf(av)));
        psum[s][j] += xv * av - sp;
      }
  }
#pragma unroll
  for (int s = 0; s < 2; ++s)
#pragma unroll
    for (int j = 0; j < 8; ++j) {
      float t = psum[s][j];
#pragma unroll
      for (int m = 1; m < 16; m <<= 1) t += __shfl_xor(t, m, 32);
      if ((lane & 15) == 0) atomicAdd(&rowsum[s * 16 + rb + j], t);
    }
  __syncthreads();
  if (tid == 0) {
    float tot = 0.0f;
    for (int r = 0; r < 32; ++r) tot += rowsum[r] + r1[m0 + r];
    atomicAdd(out, tot * (1.0f / (float)BATCH));
  }
}

extern "C" void kernel_launch(void* const* d_in, const int* in_sizes, int n_in,
                              void* d_out, int out_size, void* d_ws, size_t ws_size,
                              hipStream_t stream) {
  const float* x    = (const float*)d_in[0];
  const float* eps  = (const float*)d_in[1];
  const float* W1   = (const float*)d_in[2];
  const float* b1   = (const float*)d_in[3];
  const float* Wmu  = (const float*)d_in[4];
  const float* bmu  = (const float*)d_in[5];
  const float* Wsig = (const float*)d_in[6];
  const float* bsig = (const float*)d_in[7];
  const float* Wd1  = (const float*)d_in[8];
  const float* bd1  = (const float*)d_in[9];
  const float* Wd2  = (const float*)d_in[10];
  const float* bd2  = (const float*)d_in[11];
  float* out = (float*)d_out;

  char* ws = (char*)d_ws;
  _Float16* h    = (_Float16*)(ws + OFF_H);   /* reused as hd after GEMM2/3 */
  _Float16* z    = (_Float16*)(ws + OFF_Z);
  float*    r1   = (float*)   (ws + OFF_R1);
  _Float16* W1p  = (_Float16*)(ws + OFF_W1P);
  _Float16* Wmup = (_Float16*)(ws + OFF_WMUP);
  _Float16* Wsgp = (_Float16*)(ws + OFF_WSGP);
  _Float16* Wd1p = (_Float16*)(ws + OFF_WD1P);
  _Float16* Wd2p = (_Float16*)(ws + OFF_WD2P);

  k_zero_out<<<1, 1, 0, stream>>>(out);

  k_pack<<<(KP1 * HIDP) / 256, 256, 0, stream>>>(W1,   W1p,  DATA, HID,  KP1,  HIDP);
  k_pack<<<(HIDP * LAT)  / 256, 256, 0, stream>>>(Wmu,  Wmup, HID,  LAT,  HIDP, LAT);
  k_pack<<<(HIDP * LAT)  / 256, 256, 0, stream>>>(Wsig, Wsgp, HID,  LAT,  HIDP, LAT);
  k_pack<<<(LAT * HIDP)  / 256, 256, 0, stream>>>(Wd1,  Wd1p, LAT,  HID,  LAT,  HIDP);
  k_pack<<<(HIDP * DATA) / 256, 256, 0, stream>>>(Wd2,  Wd2p, HID,  DATA, HIDP, DATA);

  k_gemm1 <<<BATCH / 32,  256, 0, stream>>>(x, W1p, b1, h);
  k_gemm23<<<BATCH / 128, 256, 0, stream>>>(h, Wmup, Wsgp, bmu, bsig, eps, z, r1);
  k_gemm4 <<<BATCH / 16,  256, 0, stream>>>(z, Wd1p, bd1, h /* hd overwrites h */);
  k_gemm5 <<<BATCH / 32,  256, 0, stream>>>(h, Wd2p, bd2, x, r1, out);
}
